// DeltaNet_22488448762167
// MI455X (gfx1250) — matrix operand order repaired and checked by reading, not compiled, not run on hardware
//
#include <hip/hip_runtime.h>
#include <hip/hip_bf16.h>

// ---------------- problem constants ----------------
#define HID 512
#define NH 8
#define HD 64
#define KTS 4
#define BATCH 2
#define SEQ 1024
#define NROWS (BATCH * SEQ)          // 2048
#define NGATE (NH * KTS)             // 32

typedef __attribute__((ext_vector_type(16))) __bf16 v16bf;
typedef __attribute__((ext_vector_type(8)))  __bf16 v8bf;
typedef __attribute__((ext_vector_type(8)))  float  v8f;

__device__ __forceinline__ unsigned short f2bf(float f) {
    unsigned int u = __float_as_uint(f);
    unsigned int r = (u + 0x7FFFu + ((u >> 16) & 1u)) >> 16;   // RNE
    return (unsigned short)r;
}

// CDNA5 async global->LDS copy, 16B per lane, tracked by ASYNCcnt.
// %0 = per-lane LDS byte address (VGPR), %1 = per-lane 64-bit global address.
__device__ __forceinline__ void async_ld_b128(unsigned lds_off, const void* gptr) {
    asm volatile("global_load_async_to_lds_b128 %0, %1, off"
                 :: "v"(lds_off), "v"(gptr) : "memory");
}
__device__ __forceinline__ void wait_asynccnt0() {
    asm volatile("s_wait_asynccnt 0x0" ::: "memory");
}

// ---------------- 1) rmsnorm(x, in_w) -> bf16 ----------------
__global__ __launch_bounds__(256) void k_rmsnorm_cast(
    const float* __restrict__ x, const float* __restrict__ in_w,
    unsigned short* __restrict__ xn)
{
    __shared__ float s_part[8];
    int row = blockIdx.x;
    int tid = threadIdx.x, lane = tid & 31, wv = tid >> 5;
    size_t base = (size_t)row * HID;
    float v0 = x[base + tid];
    float v1 = x[base + tid + 256];
    float sq = v0 * v0 + v1 * v1;
    #pragma unroll
    for (int off = 16; off; off >>= 1) sq += __shfl_xor(sq, off, 32);
    if (lane == 0) s_part[wv] = sq;
    __syncthreads();
    float tot = 0.f;
    #pragma unroll
    for (int i = 0; i < 8; ++i) tot += s_part[i];
    float inv = rsqrtf(tot / (float)HID + 1e-5f);
    xn[base + tid]       = f2bf(v0 * inv * in_w[tid]);
    xn[base + tid + 256] = f2bf(v1 * inv * in_w[tid + 256]);
}

// ---------------- 2) generic f32 -> bf16 cast ----------------
__global__ __launch_bounds__(256) void k_cast_bf16(
    const float* __restrict__ src, unsigned short* __restrict__ dst, int n)
{
    int i = blockIdx.x * 256 + threadIdx.x;
    if (i < n) dst[i] = f2bf(src[i]);
}

// ---------------- 3) bf16 WMMA GEMM: C(MxN) = A(MxK) * W(NxK)^T ----------------
// block = 256 (8 waves), tile = 128(M) x 64(N), K-step 32.
// Double-buffered B tiles staged via global_load_async_to_lds_b128 (ASYNCcnt).
// Requires M % 128 == 0, K % 32 == 0.  N arbitrary (guarded).
__global__ __launch_bounds__(256) void k_gemm_bf16(
    const unsigned short* __restrict__ A,   // M x K row-major bf16
    const unsigned short* __restrict__ W,   // N x K row-major bf16 (B = W^T)
    float* __restrict__ C, int M, int N, int K)
{
    __shared__ __align__(64) unsigned short ldsB[2][64][32];   // 2 x 4KB
    int tid  = threadIdx.x;
    int lane = tid & 31;
    int w    = tid >> 5;
    int l16  = lane & 15;
    bool hi  = (lane >= 16);

    int m0 = blockIdx.x * 128 + w * 16;
    int n0 = blockIdx.y * 64;

    v8f acc[4];
    #pragma unroll
    for (int n = 0; n < 4; ++n)
        #pragma unroll
        for (int i = 0; i < 8; ++i) acc[n][i] = 0.f;

    const unsigned short* arow = A + (size_t)(m0 + l16) * K;
    // A fragment layout (contiguous, matches B): lanes<16 hold K {0..15},
    // lanes>=16 hold K {16..31}
    const int c0 = hi ? 16 : 0;
    const int c1 = hi ? 24 : 8;
    const int kOff = hi ? 16 : 0;           // B: lanes<16 K0..15, lanes>=16 K16..31

    // B staging assignment: thread -> (row within tile, 8-element chunk)
    int brow = tid >> 2;                    // 0..63 local B row (= output col)
    int bcol = (tid & 3) * 8;               // 0,8,16,24
    int gn = n0 + brow;
    bool valid = (gn < N);
    const unsigned short* wrow = W + (size_t)gn * K + bcol;

    unsigned ldsAddr0 = (unsigned)(size_t)&ldsB[0][brow][bcol];
    unsigned ldsAddr1 = (unsigned)(size_t)&ldsB[1][brow][bcol];

    if (!valid) {   // zero tail rows once (they are never overwritten)
        #pragma unroll
        for (int i = 0; i < 8; ++i) {
            ldsB[0][brow][bcol + i] = 0;
            ldsB[1][brow][bcol + i] = 0;
        }
    }

    // prologue: stage K-tile 0 into buffer 0
    if (valid) async_ld_b128(ldsAddr0, wrow);
    wait_asynccnt0();
    __syncthreads();

    int buf = 0;
    for (int k0 = 0; k0 < K; k0 += 32) {
        // kick off next tile into the other buffer while we compute
        if (valid && (k0 + 32 < K))
            async_ld_b128(buf ? ldsAddr0 : ldsAddr1, wrow + k0 + 32);

        if (k0 + 32 < K) __builtin_prefetch(arow + k0 + 32, 0, 1);  // global_prefetch_b8

        v8bf a_lo = *(const v8bf*)&arow[k0 + c0];
        v8bf a_hi = *(const v8bf*)&arow[k0 + c1];
        v16bf a;
        #pragma unroll
        for (int i = 0; i < 8; ++i) { a[i] = a_lo[i]; a[i + 8] = a_hi[i]; }

        #pragma unroll
        for (int n = 0; n < 4; ++n) {
            v16bf b = *(const v16bf*)&ldsB[buf][n * 16 + l16][kOff];
            acc[n] = __builtin_amdgcn_wmma_f32_16x16x32_bf16(
                false, a, false, b, (short)0, acc[n], false, false);
        }

        // next tile's async copy must have landed before anyone reads it,
        // and all reads of the current buffer must finish before it is
        // overwritten by the next iteration's copy -> one wait + one barrier.
        wait_asynccnt0();
        __syncthreads();
        buf ^= 1;
    }

    int rbase = m0 + (hi ? 8 : 0);          // C layout: lanes<16 rows 0..7, >=16 rows 8..15
    #pragma unroll
    for (int n = 0; n < 4; ++n) {
        int col = n0 + n * 16 + l16;
        if (col < N) {
            #pragma unroll
            for (int r = 0; r < 8; ++r)
                C[(size_t)(rbase + r) * N + col] = acc[n][r];
        }
    }
}

// ---------------- 4) RoPE(q,k) + rmsnorm(k)*mask + rmsnorm(v) + alpha/mix gates ----------------
__global__ __launch_bounds__(256) void k_rope_gate(
    float* __restrict__ q, float* __restrict__ k, float* __restrict__ v,
    const float* __restrict__ aLog, const float* __restrict__ mLog,
    const float* __restrict__ ab, const float* __restrict__ mb,
    const float* __restrict__ kn_w, const float* __restrict__ vn_w,
    const int* __restrict__ mask,
    float* __restrict__ alpha, float* __restrict__ mix)
{
    __shared__ float s_k2[NH], s_v2[NH];
    __shared__ float qv[HID], kv[HID];
    int row = blockIdx.x;
    int t   = row & (SEQ - 1);
    int tid = threadIdx.x;
    size_t base = (size_t)row * HID;
    const float lbs = __logf(10000.0f) / 32.0f;

    if (tid < NH) { s_k2[tid] = 0.f; s_v2[tid] = 0.f; }
    __syncthreads();

    #pragma unroll
    for (int p = 0; p < 2; ++p) {
        int e = tid + p * 256;
        int h = e >> 6, i = e & 63;
        size_t hb = base + (size_t)h * HD;
        float qe, ke;
        if (i < 32) {
            float ang = (float)t * __expf(-(float)i * lbs);
            float c = __cosf(ang), s = __sinf(ang);
            qe = q[hb + i] * c - q[hb + i + 32] * s;
            ke = k[hb + i] * c - k[hb + i + 32] * s;
        } else {
            int j = i - 32;
            float ang = (float)t * __expf(-(float)j * lbs);
            float c = __cosf(ang), s = __sinf(ang);
            qe = q[hb + j] * s + q[hb + i] * c;
            ke = k[hb + j] * s + k[hb + i] * c;
        }
        qv[e] = qe; kv[e] = ke;
        atomicAdd(&s_k2[h], ke * ke);
        float ve = v[base + e];
        atomicAdd(&s_v2[h], ve * ve);
    }
    __syncthreads();

    float mflt = (float)mask[row];
    #pragma unroll
    for (int p = 0; p < 2; ++p) {
        int e = tid + p * 256;
        int h = e >> 6, i = e & 63;
        float sk = rsqrtf(s_k2[h] / (float)HD + 1e-5f);
        float sv = rsqrtf(s_v2[h] / (float)HD + 1e-5f);
        q[base + e] = qv[e];
        k[base + e] = kv[e] * sk * kn_w[i] * mflt;   // fold mask into k
        v[base + e] = v[base + e] * sv * vn_w[i];
    }

    if (tid < NGATE) {
        float al = aLog[(size_t)row * NGATE + tid] + ab[tid];
        float s = 1.0f / (1.0f + __expf(-al));
        alpha[(size_t)row * NGATE + tid] = fminf(fmaxf(s, 1e-4f), 0.9995f);
    }
    if (tid < NH) {
        float lm[KTS], mx = -1e30f;
        #pragma unroll
        for (int j = 0; j < KTS; ++j) {
            lm[j] = mLog[(size_t)row * NGATE + tid * KTS + j] + mb[tid * KTS + j];
            mx = fmaxf(mx, lm[j]);
        }
        float sum = 0.f;
        #pragma unroll
        for (int j = 0; j < KTS; ++j) { lm[j] = __expf(lm[j] - mx); sum += lm[j]; }
        #pragma unroll
        for (int j = 0; j < KTS; ++j)
            mix[(size_t)row * NGATE + tid * KTS + j] = lm[j] / sum;
    }
}

// ---------------- 5) sequential scan: 16 WGs (b,h), 4 timescale states each ----------------
// state h[64][64] per (b,h,k); thread (kk,e) owns column e of state kk, in 64 VGPRs.
__global__ __launch_bounds__(256) void k_scan(
    const float* __restrict__ q, const float* __restrict__ k,
    const float* __restrict__ v, const float* __restrict__ alpha,
    const float* __restrict__ mix, unsigned short* __restrict__ ybf)
{
    __shared__ float s_q[HD], s_k[HD], s_v[HD], s_a[KTS], s_m[KTS];
    __shared__ float s_red[8];
    __shared__ float s_y[KTS][HD];

    int bh = blockIdx.x;
    int b = bh >> 3, h = bh & 7;
    int tid = threadIdx.x;
    int kk = tid >> 6, e = tid & 63;
    int lane = tid & 31, wv = tid >> 5;

    float hst[HD];
    #pragma unroll
    for (int d = 0; d < HD; ++d) hst[d] = 0.f;

    const float cclip = 16.0f;   // sqrt(HD) * CLIP = 8 * 2

    for (int t = 0; t < SEQ; ++t) {
        size_t rb = ((size_t)(b * SEQ + t)) * HID + (size_t)h * HD;
        size_t gb = ((size_t)(b * SEQ + t)) * NGATE + (size_t)h * KTS;
        if      (tid < 64)  s_q[tid]       = q[rb + tid];
        else if (tid < 128) s_k[tid - 64]  = k[rb + tid - 64];
        else if (tid < 192) s_v[tid - 128] = v[rb + tid - 128];
        else if (tid < 196) s_a[tid - 192] = alpha[gb + tid - 192];
        else if (tid < 200) s_m[tid - 196] = mix[gb + tid - 196];
        __syncthreads();

        float a = s_a[kk], ve = s_v[e];
        float sq = 0.f;
        #pragma unroll
        for (int d = 0; d < HD; ++d) {
            hst[d] = fmaf(a, hst[d], s_k[d] * ve);
            sq = fmaf(hst[d], hst[d], sq);
        }
        #pragma unroll
        for (int off = 16; off; off >>= 1) sq += __shfl_xor(sq, off, 32);
        if (lane == 0) s_red[wv] = sq;
        __syncthreads();
        float tot = s_red[kk * 2] + s_red[kk * 2 + 1];
        float scale = fminf(cclip / (__fsqrt_rn(tot) + 1e-6f), 1.0f);

        float y = 0.f;
        #pragma unroll
        for (int d = 0; d < HD; ++d) {
            hst[d] *= scale;
            y = fmaf(s_q[d], hst[d], y);
        }
        s_y[kk][e] = y * s_m[kk];
        __syncthreads();
        if (tid < 64) {
            float yo = s_y[0][tid] + s_y[1][tid] + s_y[2][tid] + s_y[3][tid];
            ybf[rb + tid] = f2bf(yo);
        }
        __syncthreads();
    }
}

// ---------------- 6) residual + LayerNorm ----------------
__global__ __launch_bounds__(256) void k_ln(
    const float* __restrict__ x, const float* __restrict__ proj,
    const float* __restrict__ ob, const float* __restrict__ ln_w,
    const float* __restrict__ ln_b, float* __restrict__ out)
{
    __shared__ float s_s[8], s_q2[8];
    int row = blockIdx.x;
    int tid = threadIdx.x, lane = tid & 31, wv = tid >> 5;
    size_t base = (size_t)row * HID;
    float r0 = x[base + tid]       + proj[base + tid]       + ob[tid];
    float r1 = x[base + tid + 256] + proj[base + tid + 256] + ob[tid + 256];
    float s = r0 + r1, q2 = r0 * r0 + r1 * r1;
    #pragma unroll
    for (int off = 16; off; off >>= 1) {
        s  += __shfl_xor(s, off, 32);
        q2 += __shfl_xor(q2, off, 32);
    }
    if (lane == 0) { s_s[wv] = s; s_q2[wv] = q2; }
    __syncthreads();
    float ts = 0.f, tq = 0.f;
    #pragma unroll
    for (int i = 0; i < 8; ++i) { ts += s_s[i]; tq += s_q2[i]; }
    float mu  = ts / (float)HID;
    float var = tq / (float)HID - mu * mu;
    float inv = rsqrtf(var + 1e-5f);
    out[base + tid]       = (r0 - mu) * inv * ln_w[tid]       + ln_b[tid];
    out[base + tid + 256] = (r1 - mu) * inv * ln_w[tid + 256] + ln_b[tid + 256];
}

// ---------------- launcher ----------------
extern "C" void kernel_launch(void* const* d_in, const int* in_sizes, int n_in,
                              void* d_out, int out_size, void* d_ws, size_t ws_size,
                              hipStream_t stream)
{
    const float* x    = (const float*)d_in[0];
    const float* in_w = (const float*)d_in[1];
    const float* qW   = (const float*)d_in[2];
    const float* kW   = (const float*)d_in[3];
    const float* vW   = (const float*)d_in[4];
    const float* aW   = (const float*)d_in[5];
    const float* ab   = (const float*)d_in[6];
    const float* mW   = (const float*)d_in[7];
    const float* mb   = (const float*)d_in[8];
    const float* oW   = (const float*)d_in[9];
    const float* ob   = (const float*)d_in[10];
    const float* kn_w = (const float*)d_in[11];
    const float* vn_w = (const float*)d_in[12];
    const float* ln_w = (const float*)d_in[13];
    const float* ln_b = (const float*)d_in[14];
    const int*   mask = (const int*)d_in[15];
    float* out = (float*)d_out;

    char* ws = (char*)d_ws;
    size_t off = 0;
    auto alloc = [&](size_t bytes) -> char* {
        char* p = ws + off;
        off += (bytes + 255) & ~(size_t)255;
        return p;
    };
    unsigned short* xn_bf = (unsigned short*)alloc((size_t)NROWS * HID * 2);
    unsigned short* wq_bf = (unsigned short*)alloc((size_t)HID * HID * 2);
    unsigned short* wk_bf = (unsigned short*)alloc((size_t)HID * HID * 2);
    unsigned short* wv_bf = (unsigned short*)alloc((size_t)HID * HID * 2);
    unsigned short* wo_bf = (unsigned short*)alloc((size_t)HID * HID * 2);
    unsigned short* wa_bf = (unsigned short*)alloc((size_t)NGATE * HID * 2);
    unsigned short* wm_bf = (unsigned short*)alloc((size_t)NGATE * HID * 2);
    float* qf   = (float*)alloc((size_t)NROWS * HID * 4);
    float* kf   = (float*)alloc((size_t)NROWS * HID * 4);
    float* vf   = (float*)alloc((size_t)NROWS * HID * 4);
    float* aLog = (float*)alloc((size_t)NROWS * NGATE * 4);
    float* mLog = (float*)alloc((size_t)NROWS * NGATE * 4);
    float* alp  = (float*)alloc((size_t)NROWS * NGATE * 4);
    float* mixb = (float*)alloc((size_t)NROWS * NGATE * 4);
    unsigned short* y_bf = (unsigned short*)alloc((size_t)NROWS * HID * 2);
    float* proj = (float*)alloc((size_t)NROWS * HID * 4);

    // 1) rmsnorm + bf16 cast of activations
    k_rmsnorm_cast<<<NROWS, 256, 0, stream>>>(x, in_w, xn_bf);

    // 2) weight casts
    int nW = HID * HID, nG = NGATE * HID;
    k_cast_bf16<<<(nW + 255) / 256, 256, 0, stream>>>(qW, wq_bf, nW);
    k_cast_bf16<<<(nW + 255) / 256, 256, 0, stream>>>(kW, wk_bf, nW);
    k_cast_bf16<<<(nW + 255) / 256, 256, 0, stream>>>(vW, wv_bf, nW);
    k_cast_bf16<<<(nW + 255) / 256, 256, 0, stream>>>(oW, wo_bf, nW);
    k_cast_bf16<<<(nG + 255) / 256, 256, 0, stream>>>(aW, wa_bf, nG);
    k_cast_bf16<<<(nG + 255) / 256, 256, 0, stream>>>(mW, wm_bf, nG);

    // 3) projections via WMMA GEMM
    dim3 gBig(NROWS / 128, HID / 64);
    dim3 gGate(NROWS / 128, 1);
    k_gemm_bf16<<<gBig, 256, 0, stream>>>(xn_bf, wq_bf, qf, NROWS, HID, HID);
    k_gemm_bf16<<<gBig, 256, 0, stream>>>(xn_bf, wk_bf, kf, NROWS, HID, HID);
    k_gemm_bf16<<<gBig, 256, 0, stream>>>(xn_bf, wv_bf, vf, NROWS, HID, HID);
    k_gemm_bf16<<<gGate, 256, 0, stream>>>(xn_bf, wa_bf, aLog, NROWS, NGATE, HID);
    k_gemm_bf16<<<gGate, 256, 0, stream>>>(xn_bf, wm_bf, mLog, NROWS, NGATE, HID);

    // 4) rope + head norms + gates
    k_rope_gate<<<NROWS, 256, 0, stream>>>(qf, kf, vf, aLog, mLog, ab, mb,
                                           kn_w, vn_w, mask, alp, mixb);

    // 5) recurrent scan (writes bf16 y directly)
    k_scan<<<BATCH * NH, 256, 0, stream>>>(qf, kf, vf, alp, mixb, y_bf);

    // 6) output projection
    k_gemm_bf16<<<gBig, 256, 0, stream>>>(y_bf, wo_bf, proj, NROWS, HID, HID);

    // 7) residual + layernorm
    k_ln<<<NROWS, 256, 0, stream>>>(x, proj, ob, ln_w, ln_b, out);
}
